// BevModel_73065983640184
// MI455X (gfx1250) — compile-verified, matrix operand order checked by reading, not run on hardware
//
#include <hip/hip_runtime.h>

namespace {
constexpr int B_ = 2, N_ = 6, D_ = 41, H_ = 16, W_ = 44, C_ = 64;
constexpr int X_ = 200, Y_ = 200, Z_ = 1;
constexpr int PPB = N_ * D_ * H_ * W_;     // points per batch = 173184
constexpr int NPTS = B_ * PPB;             // 346368
constexpr int DHW = D_ * H_ * W_;          // 28864
}

typedef __attribute__((ext_vector_type(16))) _Float16 v16h_t;
typedef __attribute__((ext_vector_type(8)))  float    v8f_t;

// Adjugate 3x3 inverse (exact for the identity/intrinsic matrices used here).
__device__ __forceinline__ void inv3x3(const float m[9], float o[9]) {
    float c00 =  (m[4]*m[8] - m[5]*m[7]);
    float c01 = -(m[3]*m[8] - m[5]*m[6]);
    float c02 =  (m[3]*m[7] - m[4]*m[6]);
    float det = m[0]*c00 + m[1]*c01 + m[2]*c02;
    float id  = 1.0f / det;
    o[0] = c00 * id;
    o[1] = -(m[1]*m[8] - m[2]*m[7]) * id;
    o[2] =  (m[1]*m[5] - m[2]*m[4]) * id;
    o[3] = c01 * id;
    o[4] =  (m[0]*m[8] - m[2]*m[6]) * id;
    o[5] = -(m[0]*m[5] - m[2]*m[3]) * id;
    o[6] = c02 * id;
    o[7] = -(m[0]*m[7] - m[1]*m[6]) * id;
    o[8] =  (m[0]*m[4] - m[1]*m[3]) * id;
}

// Zero-fill d_out; zeros produced on the matrix pipe (D = 0x0 + 0), 128-bit stores.
// WMMA executes before any divergence (EXEC all-1s requirement).
__global__ void __launch_bounds__(256) bev_zero_wmma(float* __restrict__ out, int n) {
    v16h_t a{};              // zero A
    v8f_t  c{};              // zero C
    c = __builtin_amdgcn_wmma_f32_16x16x32_f16(
            /*neg_a=*/false, a, /*neg_b=*/false, a,
            /*c_mod=*/(short)0, c, /*reuse_a=*/false, /*reuse_b=*/false);
    int base = (blockIdx.x * blockDim.x + threadIdx.x) * 8;
    if (base + 8 <= n) {
        float4* p = (float4*)(out + base);
        p[0] = make_float4(c[0], c[1], c[2], c[3]);
        p[1] = make_float4(c[4], c[5], c[6], c[7]);
    } else {
        for (int i = 0; i < 8; ++i)
            if (base + i < n) out[base + i] = c[i];
    }
}

// 16 threads per point; lane handles 4 consecutive channels (float4, coalesced).
// Per-(b,n) combined matrices computed once per block into LDS by threads 0..11.
__global__ void __launch_bounds__(256) bev_scatter(
        const float* __restrict__ feat,        // (B,N,D,H,W,C)
        const float* __restrict__ post_trans,  // (B,N,3)
        const float* __restrict__ post_rots,   // (B,N,3,3)
        const float* __restrict__ intrin,      // (B,N,3,3)
        const float* __restrict__ extrin,      // (B,N,4,4)
        const float* __restrict__ frustum,     // (D,H,W,3)
        const float* __restrict__ res3,        // (3)
        const float* __restrict__ start3,      // (3)
        float* __restrict__ out)               // (B,C,X,Y)
{
    __shared__ float sh[B_ * N_][24];  // [0..8]=inv(post_rots) [9..17]=rot@inv(K) [18..20]=trans [21..23]=post_trans

    const int t = threadIdx.x;
    if (t < B_ * N_) {
        float pr[9], pri[9], K[9], Ki[9], A[9], Ai[9];
        #pragma unroll
        for (int i = 0; i < 9; ++i) { pr[i] = post_rots[t*9 + i]; K[i] = intrin[t*9 + i]; }
        #pragma unroll
        for (int r = 0; r < 3; ++r)
            #pragma unroll
            for (int c = 0; c < 3; ++c) A[r*3 + c] = extrin[t*16 + r*4 + c];
        const float tx = extrin[t*16 + 3], ty = extrin[t*16 + 7], tz = extrin[t*16 + 11];
        inv3x3(pr, pri); inv3x3(K, Ki); inv3x3(A, Ai);
        #pragma unroll
        for (int i = 0; i < 9; ++i) sh[t][i] = pri[i];
        #pragma unroll
        for (int r = 0; r < 3; ++r)
            #pragma unroll
            for (int c = 0; c < 3; ++c)
                sh[t][9 + r*3 + c] = Ai[r*3+0]*Ki[c] + Ai[r*3+1]*Ki[3+c] + Ai[r*3+2]*Ki[6+c];
        sh[t][18] = -(Ai[0]*tx + Ai[1]*ty + Ai[2]*tz);
        sh[t][19] = -(Ai[3]*tx + Ai[4]*ty + Ai[5]*tz);
        sh[t][20] = -(Ai[6]*tx + Ai[7]*ty + Ai[8]*tz);
        sh[t][21] = post_trans[t*3 + 0];
        sh[t][22] = post_trans[t*3 + 1];
        sh[t][23] = post_trans[t*3 + 2];
    }
    __syncthreads();

    const unsigned gt  = blockIdx.x * blockDim.x + t;
    const unsigned pid = gt >> 4;        // point id
    const int      cl  = gt & 15;        // channel group (4 channels)
    if (pid >= (unsigned)NPTS) return;

    const int b    = (int)(pid / PPB);
    const int rem  = (int)(pid - (unsigned)b * PPB);
    const int n    = rem / DHW;
    const int fidx = rem - n * DHW;
    const float* S = sh[b * N_ + n];

    const float* fvec = feat + (size_t)pid * C_ + cl * 4;
    __builtin_prefetch(fvec, 0, 1);      // global_prefetch_b8: pull feature line early

    const float* fr = frustum + (size_t)fidx * 3;
    const float fx = fr[0], fy = fr[1], fz = fr[2];

    // pts = frustum - post_trans; pts = inv(post_rots) @ pts
    const float vx = fx - S[21], vy = fy - S[22], vz = fz - S[23];
    float ux = S[0]*vx + S[1]*vy + S[2]*vz;
    float uy = S[3]*vx + S[4]*vy + S[5]*vz;
    float uz = S[6]*vx + S[7]*vy + S[8]*vz;
    // un-project: [x*z, y*z, z]
    ux *= uz; uy *= uz;
    // (rot @ inv(K)) @ pts + trans
    const float gx = S[ 9]*ux + S[10]*uy + S[11]*uz + S[18];
    const float gy = S[12]*ux + S[13]*uy + S[14]*uz + S[19];
    const float gz = S[15]*ux + S[16]*uy + S[17]*uz + S[20];

    const float r0 = res3[0], r1 = res3[1], r2 = res3[2];
    const float s0 = start3[0] - 0.5f*r0, s1 = start3[1] - 0.5f*r1, s2 = start3[2] - 0.5f*r2;
    // C-style int cast == astype(int32): truncation toward zero (matches reference quirk)
    const int xi = (int)((gx - s0) / r0);
    const int yi = (int)((gy - s1) / r1);
    const int zi = (int)((gz - s2) / r2);
    if (xi < 0 || xi >= X_ || yi < 0 || yi >= Y_ || zi < 0 || zi >= Z_) return;

    const float4 f = *(const float4*)fvec;
    float* o = out + (((size_t)b * C_ + cl * 4) * X_ + xi) * Y_ + yi;
    atomicAdd(o,               f.x);
    atomicAdd(o +     X_ * Y_, f.y);
    atomicAdd(o + 2 * X_ * Y_, f.z);
    atomicAdd(o + 3 * X_ * Y_, f.w);
}

extern "C" void kernel_launch(void* const* d_in, const int* in_sizes, int n_in,
                              void* d_out, int out_size, void* d_ws, size_t ws_size,
                              hipStream_t stream) {
    (void)in_sizes; (void)n_in; (void)d_ws; (void)ws_size;
    const float* feat       = (const float*)d_in[0];
    const float* post_trans = (const float*)d_in[1];
    const float* post_rots  = (const float*)d_in[2];
    const float* intrin     = (const float*)d_in[3];
    const float* extrin     = (const float*)d_in[4];
    const float* frustum    = (const float*)d_in[5];
    const float* res3       = (const float*)d_in[6];
    const float* start3     = (const float*)d_in[7];
    float* out = (float*)d_out;

    // 1) zero the BEV grid (atomics accumulate into it)
    const int zblocks = (out_size + 2047) / 2048;   // 8 floats/thread, 256 threads
    bev_zero_wmma<<<zblocks, 256, 0, stream>>>(out, out_size);

    // 2) scatter-add features
    const long long total = (long long)NPTS * 16;
    const int sblocks = (int)((total + 255) / 256);
    bev_scatter<<<sblocks, 256, 0, stream>>>(feat, post_trans, post_rots, intrin,
                                             extrin, frustum, res3, start3, out);
}
